// WaveNeXtOverlap_46385646796814
// MI455X (gfx1250) — matrix operand order repaired and checked by reading, not compiled, not run on hardware
//
#include <hip/hip_runtime.h>
#include <hip/hip_bf16.h>
#include <math.h>

#define C_IN  100
#define CH    512
#define HCH   1536
#define NFFT_ 1024
#define HOP_  256
#define NL_   8
#define KOV   4
#define BB    16
#define TT    1024
#define TOK   (BB*TT)
#define LOV   (TT + KOV - 1)   // 1027

typedef __bf16 bf16;
typedef __attribute__((ext_vector_type(16))) __bf16 v16bf;
typedef __attribute__((ext_vector_type(8)))  __bf16 v8bf;
typedef __attribute__((ext_vector_type(8)))  float  v8f;

union BF16x16 { v16bf v; v8bf h[2]; };

// A fragment (16x32 bf16, M=row, K contiguous in memory, row-major A).
// lanes 0-15: K = kbase+{0..7, 16..23}; lanes 16-31: K = kbase+{8..15, 24..31}
__device__ __forceinline__ v16bf frag_A_row(const bf16* rowptr, int kbase, int lane) {
  const int kb = (lane & 16) ? 8 : 0;
  const bf16* p = rowptr + kbase + kb;
  BF16x16 u;
  u.h[0] = *(const v8bf*)(p);
  u.h[1] = *(const v8bf*)(p + 16);
  return u.v;
}

// B fragment (32x16 bf16, N=col). Weights stored [N][K] row-major so the 16
// K-values per lane are contiguous. lanes 0-15: K=kbase+{0..7,16..23}; 16-31: +{8..15,24..31}
__device__ __forceinline__ v16bf frag_B_col(const bf16* colptr, int kbase, int lane) {
  const int kb = (lane & 16) ? 8 : 0;
  const bf16* p = colptr + kbase + kb;
  BF16x16 u;
  u.h[0] = *(const v8bf*)(p);
  u.h[1] = *(const v8bf*)(p + 16);
  return u.v;
}

__device__ __forceinline__ v16bf zero16() {
  BF16x16 u;
  #pragma unroll
  for (int i = 0; i < 16; ++i) u.v[i] = (__bf16)0.0f;
  return u.v;
}

__device__ __forceinline__ v8f wmma_bf16(v16bf a, v16bf b, v8f c) {
  return __builtin_amdgcn_wmma_f32_16x16x32_bf16(false, a, false, b, (short)0, c,
                                                 false, false);
}

__device__ __forceinline__ float gelu_exact(float v) {
  return 0.5f * v * (1.0f + erff(v * 0.70710678118654752f));
}

// Software-pipelined 32x64 GEMM tile: D[32x64] = A[32xK] x W^T, W stored [N][K].
// acc[0..3] = rows row0..row0+15, acc[4..7] = rows row0+16..row0+31.
template <int KDIM>
__device__ __forceinline__ void gemm_tile_32x64(const bf16* __restrict__ A,
                                                const bf16* __restrict__ W,
                                                int row0, int colbase, int lane,
                                                v8f acc[8]) {
  const bf16* arow0 = A + (size_t)(row0 + (lane & 15)) * KDIM;
  const bf16* arow1 = arow0 + (size_t)16 * KDIM;
  const bf16* wcol[4];
  #pragma unroll
  for (int j = 0; j < 4; ++j)
    wcol[j] = W + (size_t)(colbase + j * 16 + (lane & 15)) * KDIM;

  v16bf a0 = frag_A_row(arow0, 0, lane);
  v16bf a1 = frag_A_row(arow1, 0, lane);
  v16bf b[4];
  #pragma unroll
  for (int j = 0; j < 4; ++j) b[j] = frag_B_col(wcol[j], 0, lane);

  #pragma unroll 2
  for (int kb = 0; kb < KDIM; kb += 32) {
    v16bf a0n, a1n, bn[4];
    const int kn = kb + 32;
    if (kn < KDIM) {                      // prefetch next step before computing
      a0n = frag_A_row(arow0, kn, lane);
      a1n = frag_A_row(arow1, kn, lane);
      #pragma unroll
      for (int j = 0; j < 4; ++j) bn[j] = frag_B_col(wcol[j], kn, lane);
    }
    #pragma unroll
    for (int j = 0; j < 4; ++j) {
      acc[j]     = wmma_bf16(a0, b[j], acc[j]);
      acc[4 + j] = wmma_bf16(a1, b[j], acc[4 + j]);
    }
    a0 = a0n; a1 = a1n;
    #pragma unroll
    for (int j = 0; j < 4; ++j) b[j] = bn[j];
  }
}

// ---------------- weight conversion ----------------
__global__ void cvt_f32_bf16(const float* __restrict__ s, bf16* __restrict__ d, int n) {
  int i = blockIdx.x * 256 + threadIdx.x;
  if (i < n) d[i] = (bf16)s[i];
}

// ov_w [HOP][NFFT][K] -> ovt [K][HOP][NFFT] bf16
__global__ void cvt_ovw(const float* __restrict__ s, bf16* __restrict__ d) {
  int i = blockIdx.x * 256 + threadIdx.x;           // over HOP*NFFT
  if (i >= HOP_ * NFFT_) return;
  int h = i / NFFT_, n = i % NFFT_;
  #pragma unroll
  for (int k = 0; k < KOV; ++k)
    d[((size_t)k * HOP_ + h) * NFFT_ + n] = (bf16)s[((size_t)h * NFFT_ + n) * KOV + k];
}

// ---------------- input conv (100->512, k=7, pad=3), fp32 ----------------
__global__ void in_conv_kernel(const float* __restrict__ x, const float* __restrict__ w,
                               const float* __restrict__ bias, float* __restrict__ tmp0) {
  int bc = blockIdx.x; int b = bc / CH, co = bc % CH;
  int t = blockIdx.y * 256 + threadIdx.x;
  const float* xb = x + (size_t)b * C_IN * TT;
  const float* wr = w + (size_t)co * C_IN * 7;
  float acc = bias[co];
  for (int ci = 0; ci < C_IN; ++ci) {
    #pragma unroll
    for (int j = 0; j < 7; ++j) {
      int tt = t + j - 3;
      if (tt >= 0 && tt < TT) acc += xb[(size_t)ci * TT + tt] * wr[ci * 7 + j];
    }
  }
  tmp0[((size_t)b * CH + co) * TT + t] = acc;
}

// LN over channels, src channel-major [B][CH][T] -> resid token-major [B*T][CH] fp32
__global__ void ln_in_kernel(const float* __restrict__ src, const float* __restrict__ g,
                             const float* __restrict__ bt, float* __restrict__ resid) {
  int token = blockIdx.x; int b = token / TT, t = token % TT;
  int tid = threadIdx.x;
  float v0 = src[((size_t)b * CH + tid) * TT + t];
  float v1 = src[((size_t)b * CH + tid + 256) * TT + t];
  __shared__ float ss[256], sq[256];
  ss[tid] = v0 + v1; sq[tid] = v0 * v0 + v1 * v1;
  __syncthreads();
  for (int s = 128; s > 0; s >>= 1) {
    if (tid < s) { ss[tid] += ss[tid + s]; sq[tid] += sq[tid + s]; }
    __syncthreads();
  }
  float mu = ss[0] * (1.0f / CH);
  float var = sq[0] * (1.0f / CH) - mu * mu;
  float rs = rsqrtf(var + 1e-5f);
  resid[(size_t)token * CH + tid]       = (v0 - mu) * rs * g[tid] + bt[tid];
  resid[(size_t)token * CH + tid + 256] = (v1 - mu) * rs * g[tid + 256] + bt[tid + 256];
}

// depthwise 7-tap + bias + channel-LN -> bf16 tokens [B*T][CH]
__global__ void dwln_kernel(const float* __restrict__ resid, const float* __restrict__ dw,
                            const float* __restrict__ db, const float* __restrict__ g,
                            const float* __restrict__ bt, bf16* __restrict__ Abf) {
  int token = blockIdx.x; int b = token / TT, t = token % TT;
  int tid = threadIdx.x;
  float y[2];
  #pragma unroll
  for (int u = 0; u < 2; ++u) {
    int c = tid + u * 256;
    float acc = db[c];
    #pragma unroll
    for (int j = 0; j < 7; ++j) {
      int tt = t + j - 3;
      if (tt >= 0 && tt < TT) acc += resid[((size_t)b * TT + tt) * CH + c] * dw[c * 7 + j];
    }
    y[u] = acc;
  }
  __shared__ float ss[256], sq[256];
  ss[tid] = y[0] + y[1]; sq[tid] = y[0] * y[0] + y[1] * y[1];
  __syncthreads();
  for (int s = 128; s > 0; s >>= 1) {
    if (tid < s) { ss[tid] += ss[tid + s]; sq[tid] += sq[tid + s]; }
    __syncthreads();
  }
  float mu = ss[0] * (1.0f / CH);
  float var = sq[0] * (1.0f / CH) - mu * mu;
  float rs = rsqrtf(var + 1e-5f);
  #pragma unroll
  for (int u = 0; u < 2; ++u) {
    int c = tid + u * 256;
    Abf[(size_t)token * CH + c] = (bf16)((y[u] - mu) * rs * g[c] + bt[c]);
  }
}

// final LN: resid token-major fp32 -> bf16 tokens
__global__ void lnL_kernel(const float* __restrict__ resid, const float* __restrict__ g,
                           const float* __restrict__ bt, bf16* __restrict__ Abf) {
  int token = blockIdx.x; int tid = threadIdx.x;
  float v0 = resid[(size_t)token * CH + tid];
  float v1 = resid[(size_t)token * CH + tid + 256];
  __shared__ float ss[256], sq[256];
  ss[tid] = v0 + v1; sq[tid] = v0 * v0 + v1 * v1;
  __syncthreads();
  for (int s = 128; s > 0; s >>= 1) {
    if (tid < s) { ss[tid] += ss[tid + s]; sq[tid] += sq[tid + s]; }
    __syncthreads();
  }
  float mu = ss[0] * (1.0f / CH);
  float var = sq[0] * (1.0f / CH) - mu * mu;
  float rs = rsqrtf(var + 1e-5f);
  Abf[(size_t)token * CH + tid]       = (bf16)((v0 - mu) * rs * g[tid] + bt[tid]);
  Abf[(size_t)token * CH + tid + 256] = (bf16)((v1 - mu) * rs * g[tid + 256] + bt[tid + 256]);
}

// ---------------- WMMA GEMMs ----------------
// Block = 8 waves sharing the SAME 64 columns (B fragments dedup in WGP$/L2),
// each wave takes a different 32-row slice.
// G = gelu(A[TOK,CH] x W1^T + b1). grid (TOK/256, HCH/64), block 256
__global__ void __launch_bounds__(256, 1)
gemm1_kernel(const bf16* __restrict__ A, const bf16* __restrict__ W1,
             const float* __restrict__ b1, bf16* __restrict__ G) {
  int wave = threadIdx.x >> 5, lane = threadIdx.x & 31;
  int rowbase = blockIdx.x * 256 + wave * 32;
  int colbase = blockIdx.y * 64;
  v8f acc[8] = {};
  gemm_tile_32x64<CH>(A, W1, rowbase, colbase, lane, acc);
  #pragma unroll
  for (int jj = 0; jj < 8; ++jj) {
    int col = colbase + (jj & 3) * 16 + (lane & 15);
    int rofs = (jj >> 2) * 16 + ((lane & 16) ? 8 : 0);
    float bias = b1[col];
    #pragma unroll
    for (int i = 0; i < 8; ++i) {
      int r = rowbase + rofs + i;
      G[(size_t)r * HCH + col] = (bf16)gelu_exact(acc[jj][i] + bias);
    }
  }
}

// resid += ls * (G[TOK,HCH] x W2^T + b2). grid (TOK/256, CH/64), block 256
__global__ void __launch_bounds__(256, 1)
gemm2_kernel(const bf16* __restrict__ G, const bf16* __restrict__ W2,
             const float* __restrict__ b2, const float* __restrict__ lsv,
             float* __restrict__ resid) {
  int wave = threadIdx.x >> 5, lane = threadIdx.x & 31;
  int rowbase = blockIdx.x * 256 + wave * 32;
  int colbase = blockIdx.y * 64;
  v8f acc[8] = {};
  gemm_tile_32x64<HCH>(G, W2, rowbase, colbase, lane, acc);
  #pragma unroll
  for (int jj = 0; jj < 8; ++jj) {
    int col = colbase + (jj & 3) * 16 + (lane & 15);
    int rofs = (jj >> 2) * 16 + ((lane & 16) ? 8 : 0);
    float bias = b2[col], scale = lsv[col];
    #pragma unroll
    for (int i = 0; i < 8; ++i) {
      size_t idx = (size_t)(rowbase + rofs + i) * CH + col;
      resid[idx] = resid[idx] + (acc[jj][i] + bias) * scale;
    }
  }
}

// o = A[TOK,CH] x Wout^T + out_b -> bf16 [TOK][NFFT]. grid (TOK/256, NFFT/64), block 256
__global__ void __launch_bounds__(256, 1)
gemm_out_kernel(const bf16* __restrict__ A, const bf16* __restrict__ W,
                const float* __restrict__ ob, bf16* __restrict__ o) {
  int wave = threadIdx.x >> 5, lane = threadIdx.x & 31;
  int rowbase = blockIdx.x * 256 + wave * 32;
  int colbase = blockIdx.y * 64;
  v8f acc[8] = {};
  gemm_tile_32x64<CH>(A, W, rowbase, colbase, lane, acc);
  #pragma unroll
  for (int jj = 0; jj < 8; ++jj) {
    int col = colbase + (jj & 3) * 16 + (lane & 15);
    int rofs = (jj >> 2) * 16 + ((lane & 16) ? 8 : 0);
    float bias = ob[col];
    #pragma unroll
    for (int i = 0; i < 8; ++i) {
      int r = rowbase + rofs + i;
      o[(size_t)r * NFFT_ + col] = (bf16)(acc[jj][i] + bias);
    }
  }
}

// overlap conv as 4 K-shifted GEMMs, pipelined:
// oo[b][hop][l] = sum_k sum_n o[b][l+k-3][n] * ovt[k][hop][n]
// Block = 8 waves sharing the same 32 hop columns, each wave a different l-tile.
// grid (ceil(65/8), HOP/32, B), block 256
__global__ void __launch_bounds__(256, 1)
ovconv_kernel(const bf16* __restrict__ o, const bf16* __restrict__ ovt,
              float* __restrict__ oo) {
  int wave = threadIdx.x >> 5, lane = threadIdx.x & 31;
  int ltile = blockIdx.x * 8 + wave;          // may overshoot; guarded below
  int hopbase = blockIdx.y * 32;
  int b = blockIdx.z;
  int lrow = ltile * 16 + (lane & 15);
  const bf16* ob = o + (size_t)b * TT * NFFT_;

  auto loadA = [&](int idx) -> v16bf {
    int k = idx >> 5, kb = (idx & 31) * 32;
    int t = lrow + k - 3;
    if (t < 0 || t >= TT) return zero16();
    return frag_A_row(ob + (size_t)t * NFFT_, kb, lane);
  };
  auto loadB = [&](int idx, int j) -> v16bf {
    int k = idx >> 5, kb = (idx & 31) * 32;
    const bf16* wcol = ovt + ((size_t)k * HOP_ + hopbase + j * 16 + (lane & 15)) * NFFT_;
    return frag_B_col(wcol, kb, lane);
  };

  v8f acc0 = {}, acc1 = {};
  v16bf a = loadA(0), b0 = loadB(0, 0), b1 = loadB(0, 1);
  #pragma unroll 2
  for (int idx = 0; idx < KOV * (NFFT_ / 32); ++idx) {
    v16bf an, b0n, b1n;
    if (idx + 1 < KOV * (NFFT_ / 32)) {
      an = loadA(idx + 1);
      b0n = loadB(idx + 1, 0);
      b1n = loadB(idx + 1, 1);
    }
    acc0 = wmma_bf16(a, b0, acc0);
    acc1 = wmma_bf16(a, b1, acc1);
    a = an; b0 = b0n; b1 = b1n;
  }
  #pragma unroll
  for (int i = 0; i < 8; ++i) {
    int l = ltile * 16 + i + ((lane & 16) ? 8 : 0);
    if (l < LOV) {
      oo[((size_t)b * HOP_ + hopbase + (lane & 15)) * LOV + l] = acc0[i];
      oo[((size_t)b * HOP_ + hopbase + 16 + (lane & 15)) * LOV + l] = acc1[i];
    }
  }
}

// fold: out[b][t] = sum_c oo[b][c][(t+2 + c*1023) % 1027]
__global__ void ovsum_kernel(const float* __restrict__ oo, float* __restrict__ out) {
  int i = blockIdx.x * 256 + threadIdx.x;
  if (i >= TOK) return;
  int b = i / TT, t = i % TT;
  const float* p = oo + (size_t)b * HOP_ * LOV;
  float s = 0.0f;
  for (int c = 0; c < HOP_; ++c) {
    int idx = (t + 2 + c * (HOP_ * KOV - 1)) % LOV;
    s += p[(size_t)c * LOV + idx];
  }
  out[i] = s;
}

extern "C" void kernel_launch(void* const* d_in, const int* in_sizes, int n_in,
                              void* d_out, int out_size, void* d_ws, size_t ws_size,
                              hipStream_t stream) {
  (void)in_sizes; (void)n_in; (void)out_size; (void)ws_size;
  const float* x       = (const float*)d_in[0];
  const float* in_w    = (const float*)d_in[1];
  const float* in_b    = (const float*)d_in[2];
  const float* norm_g  = (const float*)d_in[3];
  const float* norm_b  = (const float*)d_in[4];
  const float* dw_w    = (const float*)d_in[5];
  const float* dw_b    = (const float*)d_in[6];
  const float* ln_g    = (const float*)d_in[7];
  const float* ln_b    = (const float*)d_in[8];
  const float* pw1_w   = (const float*)d_in[9];
  const float* pw1_b   = (const float*)d_in[10];
  const float* pw2_w   = (const float*)d_in[11];
  const float* pw2_b   = (const float*)d_in[12];
  const float* ls      = (const float*)d_in[13];
  const float* normL_g = (const float*)d_in[14];
  const float* normL_b = (const float*)d_in[15];
  const float* out_w   = (const float*)d_in[16];
  const float* out_b   = (const float*)d_in[17];
  const float* ov_w    = (const float*)d_in[18];
  float* out = (float*)d_out;

  char* base = (char*)d_ws;
  size_t off = 0;
  auto alloc = [&](size_t bytes) -> char* {
    char* p = base + off;
    off += (bytes + 255) & ~(size_t)255;
    return p;
  };
  float* tmp0  = (float*)alloc((size_t)BB * CH * TT * 4);
  float* resid = (float*)alloc((size_t)TOK * CH * 4);
  bf16*  Abf   = (bf16*)alloc((size_t)TOK * CH * 2);
  bf16*  Gbf   = (bf16*)alloc((size_t)TOK * HCH * 2);
  bf16*  w1bf  = (bf16*)alloc((size_t)NL_ * HCH * CH * 2);
  bf16*  w2bf  = (bf16*)alloc((size_t)NL_ * CH * HCH * 2);
  bf16*  owbf  = (bf16*)alloc((size_t)NFFT_ * CH * 2);
  bf16*  ovt   = (bf16*)alloc((size_t)KOV * HOP_ * NFFT_ * 2);
  bf16*  obf   = (bf16*)alloc((size_t)TOK * NFFT_ * 2);
  float* oo    = (float*)alloc((size_t)BB * HOP_ * LOV * 4);

  // weight conversions (every call; no cached state)
  {
    int n1 = NL_ * HCH * CH;
    cvt_f32_bf16<<<(n1 + 255) / 256, 256, 0, stream>>>(pw1_w, w1bf, n1);
    int n2 = NL_ * CH * HCH;
    cvt_f32_bf16<<<(n2 + 255) / 256, 256, 0, stream>>>(pw2_w, w2bf, n2);
    int n3 = NFFT_ * CH;
    cvt_f32_bf16<<<(n3 + 255) / 256, 256, 0, stream>>>(out_w, owbf, n3);
    int n4 = HOP_ * NFFT_;
    cvt_ovw<<<(n4 + 255) / 256, 256, 0, stream>>>(ov_w, ovt);
  }

  // stem: conv + LN -> token-major residual stream
  in_conv_kernel<<<dim3(BB * CH, TT / 256), 256, 0, stream>>>(x, in_w, in_b, tmp0);
  ln_in_kernel<<<TOK, 256, 0, stream>>>(tmp0, norm_g, norm_b, resid);

  // 8 ConvNeXt blocks
  for (int l = 0; l < NL_; ++l) {
    dwln_kernel<<<TOK, 256, 0, stream>>>(resid,
        dw_w + (size_t)l * CH * 7, dw_b + (size_t)l * CH,
        ln_g + (size_t)l * CH, ln_b + (size_t)l * CH, Abf);
    gemm1_kernel<<<dim3(TOK / 256, HCH / 64), 256, 0, stream>>>(
        Abf, w1bf + (size_t)l * HCH * CH, pw1_b + (size_t)l * HCH, Gbf);
    gemm2_kernel<<<dim3(TOK / 256, CH / 64), 256, 0, stream>>>(
        Gbf, w2bf + (size_t)l * CH * HCH, pw2_b + (size_t)l * CH,
        ls + (size_t)l * CH, resid);
  }

  // final LN + 1x1 out conv
  lnL_kernel<<<TOK, 256, 0, stream>>>(resid, normL_g, normL_b, Abf);
  gemm_out_kernel<<<dim3(TOK / 256, NFFT_ / 64), 256, 0, stream>>>(Abf, owbf, out_b, obf);

  // overlap conv (WMMA) + fold
  ovconv_kernel<<<dim3(((LOV + 15) / 16 + 7) / 8, HOP_ / 32, BB), 256, 0, stream>>>(obf, ovt, oo);
  ovsum_kernel<<<(TOK + 255) / 256, 256, 0, stream>>>(oo, out);
}
